// LexicalCIRCached_27101243638187
// MI455X (gfx1250) — compile-verified, matrix operand order checked by reading, not run on hardware
//
#include <hip/hip_runtime.h>
#include <stdint.h>
#include <stddef.h>

#define B_N   256
#define V_N   30522
#define V_P   30528   // V padded to multiple of 64
#define D_N   768
#define M_N   4096
#define BOT_N 512
#define K_TOP 768
#define TEMP_INV 20.0f
#define EPS_F 1e-6f

// GEMM tiling: block = 256 thr = 8 waves; wave -> 32x32 C; block -> 128(M) x 64(N)
// K staged in steps of 64 through LDS, double-buffered.
#define KSTEP 64
#define A_TILE_BYTES (128 * KSTEP * 2)             // 16 KB
#define B_TILE_BYTES (64 * KSTEP * 2)              // 8 KB
#define STAGE_BYTES  (A_TILE_BYTES + B_TILE_BYTES) // 24 KB
#define CHUNKS_PER_THREAD 6                        // 1536 x 16B chunks / 256 thr

typedef __attribute__((ext_vector_type(16))) __bf16 v16bf;
typedef __attribute__((ext_vector_type(8)))  float  v8f;
typedef int v4i_a __attribute__((vector_size(16)));   // matches builtin param type

#ifndef __has_builtin
#define __has_builtin(x) 0
#endif
#if __has_builtin(__builtin_amdgcn_global_load_async_to_lds_b128)
#define HAVE_ASYNC_LDS 1
#else
#define HAVE_ASYNC_LDS 0
#endif
#if __has_builtin(__builtin_amdgcn_s_wait_asynccnt)
#define WAIT_ASYNC(n) __builtin_amdgcn_s_wait_asynccnt(n)
#else
#define WAIT_ASYNC(n) asm volatile("s_wait_asynccnt %0" ::"n"(n))
#endif

__device__ __forceinline__ uint16_t f2bf(float f) {
  uint32_t u = __float_as_uint(f);
  uint32_t r = u + 0x7FFFu + ((u >> 16) & 1u);   // round-to-nearest-even
  return (uint16_t)(r >> 16);
}

// ---------------------------------------------------------------------------
// Stage one K-step of A (128 rows) + B (64 rows) into LDS.
// chunk ids: [0,1024) -> A (row = cid>>3, part = cid&7), [1024,1536) -> B.
// ---------------------------------------------------------------------------
__device__ __forceinline__ void issue_stage(const uint16_t* __restrict__ A,
                                            const uint16_t* __restrict__ Bt,
                                            int K, int gm0, int gn0, int k0,
                                            char* sbase, int t) {
#pragma unroll
  for (int j = 0; j < CHUNKS_PER_THREAD; ++j) {
    const int cid  = t + 256 * j;
    const bool isB = cid >= 1024;
    const int c    = isB ? (cid - 1024) : cid;
    const int row  = c >> 3;
    const int part = c & 7;
    const uint16_t* g = (isB ? (Bt + (size_t)(gn0 + row) * K)
                             : (A  + (size_t)(gm0 + row) * K)) + k0 + part * 8;
    char* l = sbase + (isB ? A_TILE_BYTES : 0) + row * (KSTEP * 2) + part * 16;
#if HAVE_ASYNC_LDS
    __builtin_amdgcn_global_load_async_to_lds_b128(
        (__attribute__((address_space(1))) v4i_a*)(g),
        (__attribute__((address_space(3))) v4i_a*)(l), 0, 0);
#else
    *(uint4*)l = *(const uint4*)g;
#endif
  }
}

// ---------------------------------------------------------------------------
// NT GEMM: C[M,N] f32 = A[M,K]bf16 * Bt[N,K]bf16^T  via v_wmma_f32_16x16x32_bf16
// LDS double-buffered async pipeline; each wave computes 32x32 (4 WMMAs/k-chunk).
// MODE 0: +bias, exact gelu, store bf16 (Cb)
// MODE 1: +bias, softplus, store f32, zero/guard cols >= Nreal (Cf)
// MODE 2: plain store f32 (Cf)
// ---------------------------------------------------------------------------
template <int MODE>
__global__ __launch_bounds__(256) void gemm_nt(
    const uint16_t* __restrict__ A, const uint16_t* __restrict__ Bt,
    float* __restrict__ Cf, uint16_t* __restrict__ Cb,
    const float* __restrict__ bias, int Nreal, int K, int ldc)
{
  __shared__ __align__(16) char smem[2 * STAGE_BYTES];   // 48 KB
  const int t    = threadIdx.x;
  const int lane = t & 31;
  const int w    = t >> 5;
  const int mt   = w >> 1;        // 0..3
  const int nt   = w & 1;         // 0..1
  const int l15  = lane & 15;
  const int sel  = lane >> 4;
  const int gm0  = blockIdx.y * 128;
  const int gn0  = blockIdx.x * 64;

  v8f acc00 = {0.f,0.f,0.f,0.f,0.f,0.f,0.f,0.f};
  v8f acc01 = acc00, acc10 = acc00, acc11 = acc00;
  union Frag { uint4 q[2]; v16bf v; };

  const int nstage = K / KSTEP;
  issue_stage(A, Bt, K, gm0, gn0, 0, smem, t);
#if !HAVE_ASYNC_LDS
  __syncthreads();
#endif

  for (int s = 0; s < nstage; ++s) {
    char* cur = smem + (s & 1) * STAGE_BYTES;
    char* nxt = smem + ((s & 1) ^ 1) * STAGE_BYTES;
    if (s + 1 < nstage) {
      issue_stage(A, Bt, K, gm0, gn0, (s + 1) * KSTEP, nxt, t);
#if HAVE_ASYNC_LDS
      WAIT_ASYNC(CHUNKS_PER_THREAD);   // oldest stage's transfers complete
#endif
    }
#if HAVE_ASYNC_LDS
    else {
      WAIT_ASYNC(0);
    }
#endif
    __syncthreads();                   // all waves' transfers visible

#pragma unroll
    for (int kc = 0; kc < 2; ++kc) {   // two 32-wide K chunks per stage
      const int ko = kc * 64 + sel * 16;
      const char* ab0 = cur + (mt * 32 + l15) * (KSTEP * 2) + ko;
      const char* ab1 = ab0 + 16 * (KSTEP * 2);
      const char* bb0 = cur + A_TILE_BYTES + (nt * 32 + l15) * (KSTEP * 2) + ko;
      const char* bb1 = bb0 + 16 * (KSTEP * 2);
      Frag fa0, fa1, fb0, fb1;
      fa0.q[0] = *(const uint4*)(ab0);      fa0.q[1] = *(const uint4*)(ab0 + 32);
      fa1.q[0] = *(const uint4*)(ab1);      fa1.q[1] = *(const uint4*)(ab1 + 32);
      fb0.q[0] = *(const uint4*)(bb0);      fb0.q[1] = *(const uint4*)(bb0 + 32);
      fb1.q[0] = *(const uint4*)(bb1);      fb1.q[1] = *(const uint4*)(bb1 + 32);
      acc00 = __builtin_amdgcn_wmma_f32_16x16x32_bf16(false, fa0.v, false, fb0.v,
                                                      (short)0, acc00, false, false);
      acc01 = __builtin_amdgcn_wmma_f32_16x16x32_bf16(false, fa0.v, false, fb1.v,
                                                      (short)0, acc01, false, false);
      acc10 = __builtin_amdgcn_wmma_f32_16x16x32_bf16(false, fa1.v, false, fb0.v,
                                                      (short)0, acc10, false, false);
      acc11 = __builtin_amdgcn_wmma_f32_16x16x32_bf16(false, fa1.v, false, fb1.v,
                                                      (short)0, acc11, false, false);
    }
    __syncthreads();                   // done reading 'cur' before it is refilled
  }

  // epilogue: wave stores its 2x2 grid of 16x16 tiles
#pragma unroll
  for (int mi = 0; mi < 2; ++mi) {
#pragma unroll
    for (int ni = 0; ni < 2; ++ni) {
      const v8f a = (mi == 0) ? ((ni == 0) ? acc00 : acc01)
                              : ((ni == 0) ? acc10 : acc11);
      const int col   = gn0 + nt * 32 + ni * 16 + l15;
      const int rbase = gm0 + mt * 32 + mi * 16 + sel * 8;
      if (MODE == 0) {
        float b = bias[col];
#pragma unroll
        for (int i = 0; i < 8; ++i) {
          float x = a[i] + b;
          float g = 0.5f * x * (1.0f + erff(x * 0.7071067811865475f));  // exact gelu
          Cb[(size_t)(rbase + i) * ldc + col] = f2bf(g);
        }
      } else if (MODE == 1) {
        const bool ok = col < Nreal;
        float b = ok ? bias[col] : 0.f;
#pragma unroll
        for (int i = 0; i < 8; ++i) {
          float x  = a[i] + b;
          float sp = (x > 20.f) ? x : log1pf(__expf(x));                // softplus
          Cf[(size_t)(rbase + i) * ldc + col] = ok ? sp : 0.f;
        }
      } else {
#pragma unroll
        for (int i = 0; i < 8; ++i)
          Cf[(size_t)(rbase + i) * ldc + col] = a[i];
      }
    }
  }
}

// ---------------------------------------------------------------------------
// f32 [rows, C] -> bf16 [rows, Cp] row-major, zero pad
// ---------------------------------------------------------------------------
__global__ void convert_pad(const float* __restrict__ in, uint16_t* __restrict__ out,
                            int C, int Cp, int total) {
  int idx = blockIdx.x * blockDim.x + threadIdx.x;
  if (idx >= total) return;
  int r = idx / Cp, c = idx - r * Cp;
  float v = (c < C) ? in[(size_t)r * C + c] : 0.f;
  out[idx] = f2bf(v);
}

// ---------------------------------------------------------------------------
// f32 [R, C] row-major -> bf16 [Cout, Rp] row-major (transpose), zero pad
// ---------------------------------------------------------------------------
__global__ void transpose_convert(const float* __restrict__ in, uint16_t* __restrict__ out,
                                  int R, int C, int Cout, int Rp) {
  __shared__ float tile[32][33];
  int c0 = blockIdx.x * 32;
  int r0 = blockIdx.y * 32;
  for (int j = 0; j < 32; j += 8) {
    int r = r0 + threadIdx.y + j;
    int c = c0 + threadIdx.x;
    tile[threadIdx.y + j][threadIdx.x] = (r < R && c < C) ? in[(size_t)r * C + c] : 0.f;
  }
  __syncthreads();
  for (int j = 0; j < 32; j += 8) {
    int oc  = c0 + threadIdx.y + j;
    int orr = r0 + threadIdx.x;
    if (oc < Cout && orr < Rp)
      out[(size_t)oc * Rp + orr] = f2bf(tile[threadIdx.x][threadIdx.y + j]);
  }
}

// ---------------------------------------------------------------------------
// Radix-select K-th largest per row (softplus output > 0, uint order == order)
// ---------------------------------------------------------------------------
__global__ void topk_thresh(const float* __restrict__ U0, const float* __restrict__ U1,
                            float* __restrict__ thrP, float* __restrict__ thrM,
                            int V, int ldu, int K) {
  const float* U  = (blockIdx.y == 0) ? U0 : U1;
  float* thr      = (blockIdx.y == 0) ? thrP : thrM;
  const int row   = blockIdx.x;
  const float* u  = U + (size_t)row * ldu;
  __shared__ unsigned hist[256];
  __shared__ unsigned s_sel, s_rem;
  unsigned prefix = 0, remaining = (unsigned)K;
  for (int pass = 3; pass >= 0; --pass) {
    hist[threadIdx.x] = 0;
    __syncthreads();
    const int shift = pass * 8;
    for (int i = threadIdx.x; i < V; i += blockDim.x) {
      unsigned v = __float_as_uint(u[i]);
      if (pass == 3 || ((v >> (shift + 8)) == prefix))
        atomicAdd(&hist[(v >> shift) & 255u], 1u);
    }
    __syncthreads();
    if (threadIdx.x == 0) {
      unsigned acc = 0; int b = 255;
      for (; b > 0; --b) {
        unsigned h = hist[b];
        if (acc + h >= remaining) break;
        acc += h;
      }
      s_sel = (unsigned)b;
      s_rem = remaining - acc;
    }
    __syncthreads();
    prefix = (prefix << 8) | s_sel;
    remaining = s_rem;
    __syncthreads();
  }
  if (threadIdx.x == 0) thr[row] = __uint_as_float(prefix);
}

// ---------------------------------------------------------------------------
// sq = clip(sr + dsp, 0) - clip(dsm, 0); row L1(dsp)+L1(dsm); sq -> bf16 padded
// ---------------------------------------------------------------------------
__global__ void combine_kernel(const float* __restrict__ uP, const float* __restrict__ uM,
                               const float* __restrict__ sr,
                               const float* __restrict__ thrP, const float* __restrict__ thrM,
                               uint16_t* __restrict__ sqb, float* __restrict__ rowSp,
                               int V, int Vp) {
  const int row = blockIdx.x;
  const float tP = thrP[row], tM = thrM[row];
  const float* up = uP + (size_t)row * Vp;
  const float* um = uM + (size_t)row * Vp;
  const float* s  = sr + (size_t)row * V;
  uint16_t* o     = sqb + (size_t)row * Vp;
  float acc = 0.f;
  for (int i = threadIdx.x; i < Vp; i += blockDim.x) {
    float sq = 0.f;
    if (i < V) {
      float dp = up[i]; dp = (dp >= tP) ? dp : 0.f;
      float dm = um[i]; dm = (dm >= tM) ? dm : 0.f;
      acc += fabsf(dp) + fabsf(dm);
      float a = s[i] + dp; a = (a > 0.f) ? a : 0.f;
      float b = (dm > 0.f) ? dm : 0.f;
      sq = a - b;
    }
    o[i] = f2bf(sq);
  }
  __shared__ float sh[256];
  sh[threadIdx.x] = acc;
  __syncthreads();
  for (int st = 128; st > 0; st >>= 1) {
    if (threadIdx.x < st) sh[threadIdx.x] += sh[threadIdx.x + st];
    __syncthreads();
  }
  if (threadIdx.x == 0) rowSp[row] = sh[0];
}

// ---------------------------------------------------------------------------
// Per-row l2 normalize: out = x / (||x|| + eps). Safe in-place (outF == in).
// ---------------------------------------------------------------------------
__global__ void l2norm_rows(const float* __restrict__ in, float* __restrict__ outF,
                            uint16_t* __restrict__ outB, int cols) {
  const int row = blockIdx.x;
  const float* x = in + (size_t)row * cols;
  float s = 0.f;
  for (int i = threadIdx.x; i < cols; i += blockDim.x) { float v = x[i]; s += v * v; }
  __shared__ float sh[256];
  sh[threadIdx.x] = s;
  __syncthreads();
  for (int st = 128; st > 0; st >>= 1) {
    if (threadIdx.x < st) sh[threadIdx.x] += sh[threadIdx.x + st];
    __syncthreads();
  }
  const float inv = 1.0f / (sqrtf(sh[0]) + EPS_F);
  for (int i = threadIdx.x; i < cols; i += blockDim.x) {
    float v = x[i] * inv;
    if (outF) outF[(size_t)row * cols + i] = v;
    if (outB) outB[(size_t)row * cols + i] = f2bf(v);
  }
}

__global__ void dotrows(const float* __restrict__ a, const float* __restrict__ b,
                        float* __restrict__ out, int cols) {
  const int row = blockIdx.x;
  float s = 0.f;
  for (int i = threadIdx.x; i < cols; i += blockDim.x)
    s += a[(size_t)row * cols + i] * b[(size_t)row * cols + i];
  __shared__ float sh[256];
  sh[threadIdx.x] = s;
  __syncthreads();
  for (int st = 128; st > 0; st >>= 1) {
    if (threadIdx.x < st) sh[threadIdx.x] += sh[threadIdx.x + st];
    __syncthreads();
  }
  if (threadIdx.x == 0) out[row] = sh[0];
}

// InfoNCE row: logsumexp([pos, neg]/T) - pos/T
__global__ void retr_row(const float* __restrict__ scores, const float* __restrict__ pos,
                         float* __restrict__ out, int Mneg) {
  const int row = blockIdx.x;
  const float* sc = scores + (size_t)row * Mneg;
  const float l0 = pos[row] * TEMP_INV;
  float mx = l0;
  for (int i = threadIdx.x; i < Mneg; i += blockDim.x) mx = fmaxf(mx, sc[i] * TEMP_INV);
  __shared__ float sh[256];
  sh[threadIdx.x] = mx;
  __syncthreads();
  for (int st = 128; st > 0; st >>= 1) {
    if (threadIdx.x < st) sh[threadIdx.x] = fmaxf(sh[threadIdx.x], sh[threadIdx.x + st]);
    __syncthreads();
  }
  const float m = sh[0];
  __syncthreads();
  float se = (threadIdx.x == 0) ? __expf(l0 - m) : 0.f;
  for (int i = threadIdx.x; i < Mneg; i += blockDim.x) se += __expf(sc[i] * TEMP_INV - m);
  sh[threadIdx.x] = se;
  __syncthreads();
  for (int st = 128; st > 0; st >>= 1) {
    if (threadIdx.x < st) sh[threadIdx.x] += sh[threadIdx.x + st];
    __syncthreads();
  }
  if (threadIdx.x == 0) out[row] = (m + logf(sh[0])) - l0;
}

__global__ void rec_row(const float* __restrict__ zhat, const float* __restrict__ zrn,
                        float* __restrict__ out, int cols) {
  const int row = blockIdx.x;
  float s = 0.f;
  for (int i = threadIdx.x; i < cols; i += blockDim.x) {
    float d = zhat[(size_t)row * cols + i] - zrn[(size_t)row * cols + i];
    s += d * d;
  }
  __shared__ float sh[256];
  sh[threadIdx.x] = s;
  __syncthreads();
  for (int st = 128; st > 0; st >>= 1) {
    if (threadIdx.x < st) sh[threadIdx.x] += sh[threadIdx.x + st];
    __syncthreads();
  }
  if (threadIdx.x == 0) out[row] = sh[0];
}

__global__ void finalize(const float* __restrict__ retrR, const float* __restrict__ recR,
                         const float* __restrict__ rowSp, float* __restrict__ out) {
  __shared__ float sh[256];
  const int t = threadIdx.x;
  sh[t] = retrR[t];
  __syncthreads();
  for (int st = 128; st > 0; st >>= 1) { if (t < st) sh[t] += sh[t + st]; __syncthreads(); }
  float sRetr = sh[0];
  __syncthreads();
  sh[t] = recR[t];
  __syncthreads();
  for (int st = 128; st > 0; st >>= 1) { if (t < st) sh[t] += sh[t + st]; __syncthreads(); }
  float sRec = sh[0];
  __syncthreads();
  sh[t] = rowSp[t];
  __syncthreads();
  for (int st = 128; st > 0; st >>= 1) { if (t < st) sh[t] += sh[t + st]; __syncthreads(); }
  float sSp = sh[0];
  if (t == 0)
    out[0] = sRetr / (float)B_N + sRec / (float)(B_N * D_N) + 1e-4f * (sSp / (float)B_N);
}

// ---------------------------------------------------------------------------
extern "C" void kernel_launch(void* const* d_in, const int* in_sizes, int n_in,
                              void* d_out, int out_size, void* d_ws, size_t ws_size,
                              hipStream_t stream) {
  (void)in_sizes; (void)n_in; (void)out_size; (void)ws_size;
  const float* ht    = (const float*)d_in[0];
  const float* sr    = (const float*)d_in[1];
  const float* z_pos = (const float*)d_in[2];
  const float* z_neg = (const float*)d_in[3];
  const float* zr    = (const float*)d_in[4];
  const float* W1p   = (const float*)d_in[5];
  const float* b1p   = (const float*)d_in[6];
  const float* W2p   = (const float*)d_in[7];
  const float* b2p   = (const float*)d_in[8];
  const float* W1m   = (const float*)d_in[9];
  const float* b1m   = (const float*)d_in[10];
  const float* W2m   = (const float*)d_in[11];
  const float* b2m   = (const float*)d_in[12];
  const float* Wdec  = (const float*)d_in[13];

  char* ws = (char*)d_ws;
  size_t off = 0;
  auto alloc = [&](size_t bytes) -> char* {
    off = (off + 255) & ~(size_t)255;
    char* p = ws + off; off += bytes; return p;
  };
  uint16_t* htb  = (uint16_t*)alloc((size_t)B_N * V_P * 2);
  uint16_t* srb  = (uint16_t*)alloc((size_t)B_N * V_P * 2);   // reused later as sq_bf16
  uint16_t* WA   = (uint16_t*)alloc((size_t)V_P * BOT_N * 2); // W1pT then W2pT
  uint16_t* WB   = (uint16_t*)alloc((size_t)V_P * BOT_N * 2); // W1mT then W2mT
  uint16_t* WdT  = (uint16_t*)alloc((size_t)D_N * V_P * 2);
  uint16_t* hPb  = (uint16_t*)alloc((size_t)B_N * BOT_N * 2);
  uint16_t* hMb  = (uint16_t*)alloc((size_t)B_N * BOT_N * 2);
  float*    uP   = (float*)alloc((size_t)B_N * V_P * 4);
  float*    uM   = (float*)alloc((size_t)B_N * V_P * 4);
  float*    thrP = (float*)alloc(B_N * 4);
  float*    thrM = (float*)alloc(B_N * 4);
  float*    rowSp= (float*)alloc(B_N * 4);
  float*    zq   = (float*)alloc((size_t)B_N * D_N * 4);
  uint16_t* zqb  = (uint16_t*)alloc((size_t)B_N * D_N * 2);
  float*    zhat = (float*)alloc((size_t)B_N * D_N * 4);
  float*    zposn= (float*)alloc((size_t)B_N * D_N * 4);
  float*    zrn  = (float*)alloc((size_t)B_N * D_N * 4);
  uint16_t* zngb = (uint16_t*)alloc((size_t)M_N * D_N * 2);
  float*  scores = (float*)alloc((size_t)B_N * M_N * 4);
  float*    pos  = (float*)alloc(B_N * 4);
  float*   retrR = (float*)alloc(B_N * 4);
  float*    recR = (float*)alloc(B_N * 4);

  const dim3 tb(32, 8);
  const int totalBV = B_N * V_P;

  // bf16 conversions of activations
  convert_pad<<<dim3((totalBV + 255) / 256), 256, 0, stream>>>(ht, htb, V_N, V_P, totalBV);
  convert_pad<<<dim3((totalBV + 255) / 256), 256, 0, stream>>>(sr, srb, V_N, V_P, totalBV);

  // Branch GEMM1 (gelu): h = gelu(ht @ W1 + b1)   [256, 512]
  transpose_convert<<<dim3((BOT_N + 31) / 32, (V_P + 31) / 32), tb, 0, stream>>>(W1p, WA, V_N, BOT_N, BOT_N, V_P);
  gemm_nt<0><<<dim3(BOT_N / 64, B_N / 128), 256, 0, stream>>>(htb, WA, nullptr, hPb, b1p, BOT_N, V_P, BOT_N);
  transpose_convert<<<dim3((BOT_N + 31) / 32, (V_P + 31) / 32), tb, 0, stream>>>(W1m, WB, V_N, BOT_N, BOT_N, V_P);
  gemm_nt<0><<<dim3(BOT_N / 64, B_N / 128), 256, 0, stream>>>(htb, WB, nullptr, hMb, b1m, BOT_N, V_P, BOT_N);

  // Branch GEMM2 (softplus): u = softplus(h @ W2 + b2)   [256, Vp]
  transpose_convert<<<dim3((V_P + 31) / 32, (BOT_N + 31) / 32), tb, 0, stream>>>(W2p, WA, BOT_N, V_N, V_P, BOT_N);
  gemm_nt<1><<<dim3(V_P / 64, B_N / 128), 256, 0, stream>>>(hPb, WA, uP, nullptr, b2p, V_N, BOT_N, V_P);
  transpose_convert<<<dim3((V_P + 31) / 32, (BOT_N + 31) / 32), tb, 0, stream>>>(W2m, WB, BOT_N, V_N, V_P, BOT_N);
  gemm_nt<1><<<dim3(V_P / 64, B_N / 128), 256, 0, stream>>>(hMb, WB, uM, nullptr, b2m, V_N, BOT_N, V_P);

  // top-k thresholds for both branches
  topk_thresh<<<dim3(B_N, 2), 256, 0, stream>>>(uP, uM, thrP, thrM, V_N, V_P, K_TOP);

  // decoder weights, reconstruction path: zhat = norm(sr @ Wdec)
  transpose_convert<<<dim3((D_N + 31) / 32, (V_P + 31) / 32), tb, 0, stream>>>(Wdec, WdT, V_N, D_N, D_N, V_P);
  gemm_nt<2><<<dim3(D_N / 64, B_N / 128), 256, 0, stream>>>(srb, WdT, zhat, nullptr, nullptr, D_N, V_P, D_N);
  l2norm_rows<<<B_N, 256, 0, stream>>>(zhat, zhat, (uint16_t*)nullptr, D_N);

  // sq, sparse loss; srb slab reused for sq_bf16 (srb no longer needed)
  combine_kernel<<<B_N, 256, 0, stream>>>(uP, uM, sr, thrP, thrM, srb, rowSp, V_N, V_P);

  // zq = norm(norm(sq @ Wdec))
  gemm_nt<2><<<dim3(D_N / 64, B_N / 128), 256, 0, stream>>>(srb, WdT, zq, nullptr, nullptr, D_N, V_P, D_N);
  l2norm_rows<<<B_N, 256, 0, stream>>>(zq, zq, (uint16_t*)nullptr, D_N);
  l2norm_rows<<<B_N, 256, 0, stream>>>(zq, zq, zqb, D_N);

  // normalize references
  l2norm_rows<<<B_N, 256, 0, stream>>>(z_pos, zposn, (uint16_t*)nullptr, D_N);
  l2norm_rows<<<B_N, 256, 0, stream>>>(zr, zrn, (uint16_t*)nullptr, D_N);
  l2norm_rows<<<M_N, 256, 0, stream>>>(z_neg, (float*)nullptr, zngb, D_N);

  // scores
  dotrows<<<B_N, 256, 0, stream>>>(zq, zposn, pos, D_N);
  gemm_nt<2><<<dim3(M_N / 64, B_N / 128), 256, 0, stream>>>(zqb, zngb, scores, nullptr, nullptr, M_N, D_N, M_N);

  // losses
  retr_row<<<B_N, 256, 0, stream>>>(scores, pos, retrR, M_N);
  rec_row<<<B_N, 256, 0, stream>>>(zhat, zrn, recR, D_N);
  finalize<<<1, 256, 0, stream>>>(retrR, recR, rowSp, (float*)d_out);
}
